// LatentVAEBlock_49555332662010
// MI455X (gfx1250) — matrix operand order repaired and checked by reading, not compiled, not run on hardware
//
#include <hip/hip_runtime.h>
#include <hip/hip_bf16.h>
#include <cstdint>
#include <cstddef>
#include <math.h>

// ---------------- problem constants ----------------
#define BATCH 4096
#define DIN   512
#define HID   2048
#define LAT   1024

typedef __bf16 bf16;
typedef bf16  v16bf __attribute__((ext_vector_type(16)));
typedef bf16  v8bf  __attribute__((ext_vector_type(8)));
typedef float v8f   __attribute__((ext_vector_type(8)));

// ---------------- helpers ----------------
__device__ __forceinline__ float waveSum(float v) {
#pragma unroll
  for (int o = 16; o > 0; o >>= 1) v += __shfl_down(v, o, 32);
  return v;
}

// CDNA5 async global->LDS copy (16B per lane), tracked by ASYNCcnt.
// lds = wave-relative LDS byte address (low 32 bits of generic pointer),
// g   = per-lane 64-bit global address.
__device__ __forceinline__ void async_b128(uint32_t lds, const void* g) {
  asm volatile("global_load_async_to_lds_b128 %0, %1, off"
               :: "v"(lds), "v"(g)
               : "memory");
}

// ---------------- fp32 -> bf16 elementwise ----------------
__global__ void cvt_bf16(const float* __restrict__ src, bf16* __restrict__ dst, int n) {
  int i = blockIdx.x * blockDim.x + threadIdx.x;
  int stride = gridDim.x * blockDim.x;
  for (; i < n; i += stride) dst[i] = (bf16)src[i];
}

// ---------------- fp32 [K x N] -> bf16 [N x K] transpose-convert ----------------
// block (32,8), grid (N/32, K/32); dims are multiples of 32
__global__ void transpose_cvt(const float* __restrict__ W, bf16* __restrict__ Wt,
                              int K, int N) {
  __shared__ float tile[32][33];
  const int bx = blockIdx.x, by = blockIdx.y;
  const int tx = threadIdx.x, ty = threadIdx.y;
#pragma unroll
  for (int j = 0; j < 32; j += 8)
    tile[ty + j][tx] = W[(size_t)(by * 32 + ty + j) * N + bx * 32 + tx];
  __syncthreads();
#pragma unroll
  for (int j = 0; j < 32; j += 8)
    Wt[(size_t)(bx * 32 + ty + j) * K + by * 32 + tx] = (bf16)tile[tx][ty + j];
}

// ---------------- WMMA GEMM: C[M,N] = A[M,K](bf16) * Wt[N,K](bf16)^T + bias ----------
// block = 256 threads (8 wave32), block tile 128x128, K-step 32
// wave grid 2(M) x 4(N); wave tile 64x32 = 4x2 fragments of 16x16
// Double-buffered LDS filled by GLOBAL_LOAD_ASYNC_TO_LDS_B128 (ASYNCcnt pipeline).
#define LDS_BUF_BYTES (128 * 32 * 2)

__global__ __launch_bounds__(256) void gemm_bf16(
    const bf16* __restrict__ A,   // M x K row-major
    const bf16* __restrict__ Bt,  // N x K row-major (transposed weight)
    const float* __restrict__ bias,  // N
    float* __restrict__ C,        // M x N
    int M, int N, int K) {
  __shared__ __align__(64) bf16 As[2][128 * 32];
  __shared__ __align__(64) bf16 Bs[2][128 * 32];

  const int tid  = threadIdx.x;
  const int lane = tid & 31;
  const int wave = tid >> 5;
  const int wm   = wave & 1;   // 0..1  -> 64 rows each
  const int wn   = wave >> 1;  // 0..3  -> 32 cols each
  const int l16  = lane & 15;
  const int half = lane >> 4;
  const int bm = blockIdx.y * 128;
  const int bn = blockIdx.x * 128;

  // Each thread stages 4 x 16B per tile: rows (row, row+64) of A and Bt.
  const int row  = tid >> 2;
  const int col8 = (tid & 3) << 3;
  const bf16* gA0 = A  + (size_t)(bm + row) * K + col8;
  const bf16* gA1 = A  + (size_t)(bm + row + 64) * K + col8;
  const bf16* gB0 = Bt + (size_t)(bn + row) * K + col8;
  const bf16* gB1 = Bt + (size_t)(bn + row + 64) * K + col8;
  const uint32_t lA0 = (uint32_t)(uintptr_t)&As[0][row * 32 + col8];
  const uint32_t lA1 = (uint32_t)(uintptr_t)&As[0][(row + 64) * 32 + col8];
  const uint32_t lB0 = (uint32_t)(uintptr_t)&Bs[0][row * 32 + col8];
  const uint32_t lB1 = (uint32_t)(uintptr_t)&Bs[0][(row + 64) * 32 + col8];

  v8f acc[4][2];
#pragma unroll
  for (int i = 0; i < 4; i++)
#pragma unroll
    for (int j = 0; j < 2; j++)
#pragma unroll
      for (int e = 0; e < 8; e++) acc[i][j][e] = 0.0f;

  const int T = K >> 5;  // K/32 tiles

  auto issue = [&](int t, int buf) {
    const uint32_t bo = buf ? (uint32_t)LDS_BUF_BYTES : 0u;
    const size_t ko = (size_t)t * 32;
    async_b128(lA0 + bo, gA0 + ko);
    async_b128(lA1 + bo, gA1 + ko);
    async_b128(lB0 + bo, gB0 + ko);
    async_b128(lB1 + bo, gB1 + ko);
  };

  issue(0, 0);  // prologue

  for (int t = 0; t < T; t++) {
    const int cur = t & 1;
    // own tile-t async loads complete:
    asm volatile("s_wait_asynccnt 0x0" ::: "memory");
    // all waves' tile-t loads complete AND all waves done reading buf cur^1:
    __syncthreads();
    if (t + 1 < T) issue(t + 1, cur ^ 1);  // overlap next tile with compute

    const bf16* as = &As[cur][0];
    const bf16* bs = &Bs[cur][0];

    // --- B fragments: lane holds col n=l16, K kb..kb+15 contiguous (32B) ---
    v16bf bfrag[2];
#pragma unroll
    for (int nf = 0; nf < 2; nf++) {
      const bf16* p = &bs[(wn * 32 + nf * 16 + l16) * 32 + half * 16];
      bfrag[nf] = *(const v16bf*)p;
    }
    // --- A fragments: lane holds row m, K = half*16+(0..15) contiguous ---
#pragma unroll
    for (int mf = 0; mf < 4; mf++) {
      const bf16* p1 = &as[(wm * 64 + mf * 16 + l16) * 32 + half * 16];
      v8bf lo = *(const v8bf*)p1;
      v8bf hi = *(const v8bf*)(p1 + 8);
      v16bf afrag = __builtin_shufflevector(lo, hi, 0, 1, 2, 3, 4, 5, 6, 7, 8,
                                            9, 10, 11, 12, 13, 14, 15);
      acc[mf][0] = __builtin_amdgcn_wmma_f32_16x16x32_bf16(
          false, afrag, false, bfrag[0], (short)0, acc[mf][0], false, false);
      acc[mf][1] = __builtin_amdgcn_wmma_f32_16x16x32_bf16(
          false, afrag, false, bfrag[1], (short)0, acc[mf][1], false, false);
    }
  }

  // --- epilogue: C element (VGPR e): row = base + half*8 + e, col = base + l16 ---
#pragma unroll
  for (int mf = 0; mf < 4; mf++)
#pragma unroll
    for (int nf = 0; nf < 2; nf++) {
      const int col  = bn + wn * 32 + nf * 16 + l16;
      const int row0 = bm + wm * 64 + mf * 16 + half * 8;
      const float bv = bias[col];
#pragma unroll
      for (int e = 0; e < 8; e++)
        C[(size_t)(row0 + e) * N + col] = acc[mf][nf][e] + bv;
    }
}

// ---------------- fused LayerNorm + SiLU, fp32 in -> bf16 out (cols = 2048) --------
__global__ __launch_bounds__(256) void ln_silu(const float* __restrict__ h,
                                               const float* __restrict__ g,
                                               const float* __restrict__ b,
                                               bf16* __restrict__ out) {
  const int row = blockIdx.x;
  const int tid = threadIdx.x;
  const float* hr = h + (size_t)row * HID;
  float lv[8], s = 0.f, ss = 0.f;
#pragma unroll
  for (int j = 0; j < 8; j++) {
    float v = hr[tid + j * 256];
    lv[j] = v; s += v; ss += v * v;
  }
  __shared__ float sm[8], sm2[8];
  float a = waveSum(s), a2 = waveSum(ss);
  if ((tid & 31) == 0) { sm[tid >> 5] = a; sm2[tid >> 5] = a2; }
  __syncthreads();
  __shared__ float mean_s, rstd_s;
  if (tid == 0) {
    float S = 0.f, SS = 0.f;
    for (int i = 0; i < 8; i++) { S += sm[i]; SS += sm2[i]; }
    float m = S * (1.0f / HID);
    float var = SS * (1.0f / HID) - m * m;
    mean_s = m;
    rstd_s = rsqrtf(var + 1e-5f);
  }
  __syncthreads();
  const float m = mean_s, r = rstd_s;
  bf16* orow = out + (size_t)row * HID;
#pragma unroll
  for (int j = 0; j < 8; j++) {
    const int c = tid + j * 256;
    float y = (lv[j] - m) * r * g[c] + b[c];
    float sg = 1.f / (1.f + __expf(-y));
    orow[c] = (bf16)(y * sg);
  }
}

// ---------------- reparameterize + KL: enc_out[4096,2048] -> z, zb, kl -------------
__global__ __launch_bounds__(256) void sample_kl(const float* __restrict__ enc_out,
                                                 const float* __restrict__ noise,
                                                 float* __restrict__ z,
                                                 bf16* __restrict__ zb,
                                                 float* __restrict__ kl_accum) {
  const int row = blockIdx.x;
  const int tid = threadIdx.x;
  const float* er = enc_out + (size_t)row * (2 * LAT);
  const float* nr = noise + (size_t)row * LAT;
  float kl = 0.f;
#pragma unroll
  for (int j = 0; j < 4; j++) {
    const int c = tid + j * 256;
    float mu = er[c];
    float lvr = er[LAT + c];
    float sp = (lvr > 20.f) ? lvr : log1pf(__expf(lvr));
    float scale = sp + 1e-5f;
    float zz = mu + scale * nr[c];
    z[(size_t)row * LAT + c] = zz;
    zb[(size_t)row * LAT + c] = (bf16)zz;
    kl += 0.5f * (scale * scale + mu * mu - 1.f) - __logf(scale);
  }
  __shared__ float red[8];
  float t = waveSum(kl);
  if ((tid & 31) == 0) red[tid >> 5] = t;
  __syncthreads();
  if (tid == 0) {
    float s = 0.f;
    for (int i = 0; i < 8; i++) s += red[i];
    atomicAdd(kl_accum, s * (1.0f / BATCH));
  }
}

// ---------------- recon loss: mean((x_hat - x)^2) ----------------
__global__ __launch_bounds__(256) void recon_loss(const float* __restrict__ xhat,
                                                  const float* __restrict__ x,
                                                  float* __restrict__ accum, int n) {
  float s = 0.f;
  for (int i = blockIdx.x * 256 + threadIdx.x; i < n; i += gridDim.x * 256) {
    float d = xhat[i] - x[i];
    s += d * d;
  }
  __shared__ float red[8];
  float t = waveSum(s);
  if ((threadIdx.x & 31) == 0) red[threadIdx.x >> 5] = t;
  __syncthreads();
  if (threadIdx.x == 0) {
    float tot = 0.f;
    for (int i = 0; i < 8; i++) tot += red[i];
    atomicAdd(accum, tot * (1.0f / ((float)BATCH * (float)DIN)));
  }
}

// ---------------- host ----------------
extern "C" void kernel_launch(void* const* d_in, const int* in_sizes, int n_in,
                              void* d_out, int out_size, void* d_ws, size_t ws_size,
                              hipStream_t stream) {
  const float* x        = (const float*)d_in[0];
  const float* noise    = (const float*)d_in[1];
  const float* enc_w_in = (const float*)d_in[2];
  const float* enc_b_in = (const float*)d_in[3];
  const float* enc_w_hid= (const float*)d_in[4];
  const float* enc_b_hid= (const float*)d_in[5];
  const float* enc_ln_g = (const float*)d_in[6];
  const float* enc_ln_b = (const float*)d_in[7];
  const float* enc_w_out= (const float*)d_in[8];
  const float* enc_b_out= (const float*)d_in[9];
  const float* dec_w_in = (const float*)d_in[10];
  const float* dec_b_in = (const float*)d_in[11];
  const float* dec_w_hid= (const float*)d_in[12];
  const float* dec_b_hid= (const float*)d_in[13];
  const float* dec_ln_g = (const float*)d_in[14];
  const float* dec_ln_b = (const float*)d_in[15];
  const float* dec_w_out= (const float*)d_in[16];
  const float* dec_b_out= (const float*)d_in[17];

  float* out    = (float*)d_out;
  float* z_out  = out;                                   // 4096*1024
  float* xhat   = out + (size_t)BATCH * LAT;             // 4096*512
  float* losses = xhat + (size_t)BATCH * DIN;            // [recon, kl]

  // workspace carve (256B aligned)
  char* wsp = (char*)d_ws;
  auto take = [&](size_t bytes) -> char* {
    char* p = wsp;
    wsp += (bytes + 255) & ~(size_t)255;
    return p;
  };
  bf16* xb        = (bf16*)take((size_t)BATCH * DIN * 2);
  bf16* wt_enc_in = (bf16*)take((size_t)HID * DIN * 2);
  bf16* wt_enc_h0 = (bf16*)take((size_t)HID * HID * 2);
  bf16* wt_enc_h1 = (bf16*)take((size_t)HID * HID * 2);
  bf16* wt_enc_out= (bf16*)take((size_t)(2 * LAT) * HID * 2);
  bf16* wt_dec_in = (bf16*)take((size_t)HID * LAT * 2);
  bf16* wt_dec_h0 = (bf16*)take((size_t)HID * HID * 2);
  bf16* wt_dec_h1 = (bf16*)take((size_t)HID * HID * 2);
  bf16* wt_dec_out= (bf16*)take((size_t)DIN * HID * 2);
  float* h        = (float*)take((size_t)BATCH * HID * 4);
  bf16* act       = (bf16*)take((size_t)BATCH * HID * 2);
  bf16* zb        = (bf16*)take((size_t)BATCH * LAT * 2);

  const dim3 tb(32, 8);
  // weight/input conversion (runs every call; ~150MB total, negligible vs GEMMs)
  cvt_bf16<<<2048, 256, 0, stream>>>(x, xb, BATCH * DIN);
  transpose_cvt<<<dim3(HID / 32, DIN / 32), tb, 0, stream>>>(enc_w_in, wt_enc_in, DIN, HID);
  transpose_cvt<<<dim3(HID / 32, HID / 32), tb, 0, stream>>>(enc_w_hid, wt_enc_h0, HID, HID);
  transpose_cvt<<<dim3(HID / 32, HID / 32), tb, 0, stream>>>(enc_w_hid + (size_t)HID * HID, wt_enc_h1, HID, HID);
  transpose_cvt<<<dim3((2 * LAT) / 32, HID / 32), tb, 0, stream>>>(enc_w_out, wt_enc_out, HID, 2 * LAT);
  transpose_cvt<<<dim3(HID / 32, LAT / 32), tb, 0, stream>>>(dec_w_in, wt_dec_in, LAT, HID);
  transpose_cvt<<<dim3(HID / 32, HID / 32), tb, 0, stream>>>(dec_w_hid, wt_dec_h0, HID, HID);
  transpose_cvt<<<dim3(HID / 32, HID / 32), tb, 0, stream>>>(dec_w_hid + (size_t)HID * HID, wt_dec_h1, HID, HID);
  transpose_cvt<<<dim3(DIN / 32, HID / 32), tb, 0, stream>>>(dec_w_out, wt_dec_out, HID, DIN);

  hipMemsetAsync(losses, 0, 2 * sizeof(float), stream);

  // ---- encoder ----
  gemm_bf16<<<dim3(HID / 128, BATCH / 128), 256, 0, stream>>>(xb, wt_enc_in, enc_b_in, h, BATCH, HID, DIN);
  ln_silu<<<BATCH, 256, 0, stream>>>(h, enc_ln_g, enc_ln_b, act);
  gemm_bf16<<<dim3(HID / 128, BATCH / 128), 256, 0, stream>>>(act, wt_enc_h0, enc_b_hid, h, BATCH, HID, HID);
  ln_silu<<<BATCH, 256, 0, stream>>>(h, enc_ln_g + HID, enc_ln_b + HID, act);
  gemm_bf16<<<dim3(HID / 128, BATCH / 128), 256, 0, stream>>>(act, wt_enc_h1, enc_b_hid + HID, h, BATCH, HID, HID);
  ln_silu<<<BATCH, 256, 0, stream>>>(h, enc_ln_g + 2 * HID, enc_ln_b + 2 * HID, act);
  gemm_bf16<<<dim3((2 * LAT) / 128, BATCH / 128), 256, 0, stream>>>(act, wt_enc_out, enc_b_out, h, BATCH, 2 * LAT, HID);

  // ---- sample + KL ----
  sample_kl<<<BATCH, 256, 0, stream>>>(h, noise, z_out, zb, losses + 1);

  // ---- decoder ----
  gemm_bf16<<<dim3(HID / 128, BATCH / 128), 256, 0, stream>>>(zb, wt_dec_in, dec_b_in, h, BATCH, HID, LAT);
  ln_silu<<<BATCH, 256, 0, stream>>>(h, dec_ln_g, dec_ln_b, act);
  gemm_bf16<<<dim3(HID / 128, BATCH / 128), 256, 0, stream>>>(act, wt_dec_h0, dec_b_hid, h, BATCH, HID, HID);
  ln_silu<<<BATCH, 256, 0, stream>>>(h, dec_ln_g + HID, dec_ln_b + HID, act);
  gemm_bf16<<<dim3(HID / 128, BATCH / 128), 256, 0, stream>>>(act, wt_dec_h1, dec_b_hid + HID, h, BATCH, HID, HID);
  ln_silu<<<BATCH, 256, 0, stream>>>(h, dec_ln_g + 2 * HID, dec_ln_b + 2 * HID, act);
  gemm_bf16<<<dim3(DIN / 128, BATCH / 128), 256, 0, stream>>>(act, wt_dec_out, dec_b_out, xhat, BATCH, DIN, HID);

  // ---- losses ----
  recon_loss<<<1024, 256, 0, stream>>>(xhat, x, losses, BATCH * DIN);
}
